// LlamaAttention_40956808135364
// MI455X (gfx1250) — compile-verified, matrix operand order checked by reading, not run on hardware
//
#include <hip/hip_runtime.h>

// ---------------------------------------------------------------------------
// LlamaAttention forward for MI455X (gfx1250, wave32, WMMA).
// Compute-bound (~206 GFLOP vs ~300MB HBM): bf16 v_wmma_f32_16x16x32_bf16 for
// all matmuls, f32 accumulate. One-time f32->bf16 conversion pass so all GEMM
// staging is pure b128 copies (no per-iteration convert VALU).
// ---------------------------------------------------------------------------

#define SEQ 2048
#define HIDDEN 2048
#define NH 32
#define HD 64
#define BATCH 2

typedef __attribute__((ext_vector_type(16))) __bf16 bf16x16;
typedef __attribute__((ext_vector_type(8)))  float  f32x8;

union Frag {
    bf16x16 v;
    uint4   q[2];
};

__device__ __forceinline__ unsigned short f2bf(float f) {
    union { float f; unsigned u; } x;
    x.f = f;
    unsigned u = x.u + 0x7FFFu + ((x.u >> 16) & 1u);   // round-to-nearest-even
    return (unsigned short)(u >> 16);
}

__device__ __forceinline__ f32x8 zero8() {
    f32x8 z = {0.f, 0.f, 0.f, 0.f, 0.f, 0.f, 0.f, 0.f};
    return z;
}

__device__ __forceinline__ f32x8 wmma_bf16(bf16x16 a, bf16x16 b, f32x8 c) {
    // (neg_a, A, neg_b, B, c_mod, C, reuse_a, reuse_b)
    return __builtin_amdgcn_wmma_f32_16x16x32_bf16(false, a, false, b,
                                                   (short)0, c, false, false);
}

// ---------------------------------------------------------------------------
// One-time f32 -> bf16 conversion (8 elements / thread, b128 in, b128 out).
// ---------------------------------------------------------------------------
__global__ __launch_bounds__(256) void cvt_f32_bf16(const float* __restrict__ in,
                                                    unsigned short* __restrict__ out) {
    size_t i = (size_t)blockIdx.x * 256 + threadIdx.x;   // one uint4 (8 bf16)
    const float4* ip = (const float4*)in;
    float4 a = ip[i * 2];
    float4 b = ip[i * 2 + 1];
    uint4 o;
    o.x = (unsigned)f2bf(a.x) | ((unsigned)f2bf(a.y) << 16);
    o.y = (unsigned)f2bf(a.z) | ((unsigned)f2bf(a.w) << 16);
    o.z = (unsigned)f2bf(b.x) | ((unsigned)f2bf(b.y) << 16);
    o.w = (unsigned)f2bf(b.z) | ((unsigned)f2bf(b.w) << 16);
    ((uint4*)out)[i] = o;
}

// ---------------------------------------------------------------------------
// GEMM: C[M=4096, N=2048] = A_bf16[4096,2048] * W_bf16[2048,2048]^T (NT: both
// operands read by rows with contiguous K -> identical A/B fragment loads).
// Block: 256 thr = 8 waves, tile 128(M) x 128(N), K-step 64.
// Wave: 32(M) x 64(N) = 2x4 WMMA tiles x 2 k-steps = 16 WMMA / iteration.
// OUT_BHSD: bf16 [B,H,S,D] (Q/K/V) vs f32 row-major [B,S,HID] (final out).
// ---------------------------------------------------------------------------
#define LDT 72   // LDS row stride in bf16 (144B: 16B aligned, bank spread)

template <bool OUT_BHSD>
__global__ __launch_bounds__(256) void gemm_nt(const unsigned short* __restrict__ A,
                                               const unsigned short* __restrict__ W,
                                               void* __restrict__ Cp) {
    __shared__ __align__(16) unsigned short As[128 * LDT];
    __shared__ __align__(16) unsigned short Bs[128 * LDT];

    const int tid  = threadIdx.x;
    const int lane = tid & 31, w = tid >> 5;
    const int g = lane >> 4, ln = lane & 15;
    const int mBase = blockIdx.y * 128;
    const int nBase = blockIdx.x * 128;
    const int waveM = (w & 3) * 32;
    const int waveN = (w >> 2) * 64;

    f32x8 acc[2][4];
#pragma unroll
    for (int ti = 0; ti < 2; ++ti)
#pragma unroll
        for (int tj = 0; tj < 4; ++tj) acc[ti][tj] = zero8();

    for (int k0 = 0; k0 < HIDDEN; k0 += 64) {
        __syncthreads();
        // ---- stage A and W tiles (128 x 64 bf16 each): pure b128 copies ----
#pragma unroll
        for (int i = 0; i < 4; ++i) {
            int idx = tid + i * 256;            // 0..1023 uint4 chunks
            int row = idx >> 3, c8 = idx & 7;
            *(uint4*)&As[row * LDT + c8 * 8] =
                *(const uint4*)&A[(size_t)(mBase + row) * HIDDEN + k0 + c8 * 8];
            *(uint4*)&Bs[row * LDT + c8 * 8] =
                *(const uint4*)&W[(size_t)(nBase + row) * HIDDEN + k0 + c8 * 8];
        }
        __syncthreads();

        // ---- fragments: lane L%16 = row, chunks at k = g*8 and g*8+16 ----
#pragma unroll
        for (int ks = 0; ks < 2; ++ks) {
            Frag fa[2], fb[4];
#pragma unroll
            for (int ti = 0; ti < 2; ++ti) {
                const unsigned short* p = &As[(waveM + 16 * ti + ln) * LDT + ks * 32 + g * 8];
                fa[ti].q[0] = *(const uint4*)p;
                fa[ti].q[1] = *(const uint4*)(p + 16);
            }
#pragma unroll
            for (int tj = 0; tj < 4; ++tj) {
                const unsigned short* p = &Bs[(waveN + 16 * tj + ln) * LDT + ks * 32 + g * 8];
                fb[tj].q[0] = *(const uint4*)p;
                fb[tj].q[1] = *(const uint4*)(p + 16);
            }
#pragma unroll
            for (int ti = 0; ti < 2; ++ti)
#pragma unroll
                for (int tj = 0; tj < 4; ++tj)
                    acc[ti][tj] = wmma_bf16(fa[ti].v, fb[tj].v, acc[ti][tj]);
        }
    }

    // ---- store: C layout = (row r+8g, col L%16) per VGPR r ----
#pragma unroll
    for (int ti = 0; ti < 2; ++ti)
#pragma unroll
        for (int tj = 0; tj < 4; ++tj)
#pragma unroll
            for (int r = 0; r < 8; ++r) {
                int row = mBase + waveM + 16 * ti + r + 8 * g;   // b*S+s
                int col = nBase + waveN + 16 * tj + ln;          // h*D+d
                float val = acc[ti][tj][r];
                if constexpr (OUT_BHSD) {
                    int b = row >> 11, s = row & (SEQ - 1);
                    int h = col >> 6, d = col & (HD - 1);
                    ((unsigned short*)Cp)[(((size_t)b * NH + h) * SEQ + s) * HD + d] = f2bf(val);
                } else {
                    ((float*)Cp)[(size_t)row * HIDDEN + col] = val;
                }
            }
}

// ---------------------------------------------------------------------------
// Flash attention: grid (S/64, B*H), 128 thr = 4 waves, each wave owns 16 Q
// rows. 64-key blocks: S = Q*K^T (4 n-tiles x 2 k-steps = 8 WMMA), online
// softmax (shfl_xor within 16-lane halves), ctx += P*V (4 d-tiles x 2 k-steps
// = 8 WMMA). P converted C-layout -> A-fragment layout via per-wave LDS.
// ---------------------------------------------------------------------------
__global__ __launch_bounds__(128) void attn_kernel(const unsigned short* __restrict__ qw,
                                                   const unsigned short* __restrict__ kw,
                                                   const unsigned short* __restrict__ vw,
                                                   const float* __restrict__ maskp,
                                                   unsigned short* __restrict__ ctx) {
    __shared__ __align__(16) unsigned short Ks[64 * LDT];      // K block, row-major [k][d]
    __shared__ __align__(16) unsigned short Vt[64 * LDT];      // V block, transposed [d][k]
    __shared__ __align__(16) unsigned short Ps[4 * 16 * LDT];  // per-wave P scratch

    const int tid  = threadIdx.x;
    const int lane = tid & 31, w = tid >> 5;
    const int g = lane >> 4, ln = lane & 15;
    const int qb = blockIdx.x * 64;
    const int bh = blockIdx.y;
    const int b = bh >> 5, h = bh & 31;

    // Q fragments for this wave's 16 rows (d = 0..63 -> 2 WMMA k-steps)
    Frag qf[2];
    {
        const size_t qbase = ((size_t)bh * SEQ + (qb + 16 * w + ln)) * HD;
#pragma unroll
        for (int ks = 0; ks < 2; ++ks) {
            const unsigned short* p = qw + qbase + ks * 32 + g * 8;
            qf[ks].q[0] = *(const uint4*)p;
            qf[ks].q[1] = *(const uint4*)(p + 16);
        }
    }

    float row_max[8], row_sum[8];
    f32x8 acc[4];
#pragma unroll
    for (int r = 0; r < 8; ++r) { row_max[r] = -1e30f; row_sum[r] = 0.f; }
#pragma unroll
    for (int t = 0; t < 4; ++t) acc[t] = zero8();

    for (int kb = 0; kb < SEQ / 64; ++kb) {
        __syncthreads();
        // stage K block 64x64 row-major (512 uint4 chunks / 128 threads)
#pragma unroll
        for (int i = 0; i < 4; ++i) {
            int idx = tid + i * 128;
            int row = idx >> 3, c = idx & 7;
            *(uint4*)&Ks[row * LDT + c * 8] =
                *(const uint4*)&kw[((size_t)bh * SEQ + kb * 64 + row) * HD + c * 8];
        }
        // stage V block transposed: Vt[d][k]
#pragma unroll
        for (int i = 0; i < 32; ++i) {
            int e = tid + i * 128;       // 0..4095 elements
            int d = e & 63, kr = e >> 6;
            Vt[d * LDT + kr] = vw[((size_t)bh * SEQ + kb * 64 + kr) * HD + d];
        }
        __syncthreads();

        // scores: 16x64 = four 16x16 tiles, accumulate over d (2 k-steps)
        f32x8 sc4[4];
#pragma unroll
        for (int t = 0; t < 4; ++t) sc4[t] = zero8();
#pragma unroll
        for (int ks = 0; ks < 2; ++ks) {
#pragma unroll
            for (int t = 0; t < 4; ++t) {
                Frag kf;
                const unsigned short* p = &Ks[(16 * t + ln) * LDT + ks * 32 + g * 8];
                kf.q[0] = *(const uint4*)p;
                kf.q[1] = *(const uint4*)(p + 16);
                sc4[t] = wmma_bf16(qf[ks].v, kf.v, sc4[t]);
            }
        }

        float mk[4];
#pragma unroll
        for (int t = 0; t < 4; ++t)
            mk[t] = maskp[(size_t)b * SEQ + kb * 64 + 16 * t + ln];

        // online softmax: row r+8g lives in half-wave g, cols across 16 lanes
#pragma unroll
        for (int r = 0; r < 8; ++r) {
            float v0 = sc4[0][r] * 0.125f + mk[0];   // 1/sqrt(64)
            float v1 = sc4[1][r] * 0.125f + mk[1];
            float v2 = sc4[2][r] * 0.125f + mk[2];
            float v3 = sc4[3][r] * 0.125f + mk[3];
            float m = fmaxf(fmaxf(v0, v1), fmaxf(v2, v3));
            m = fmaxf(m, __shfl_xor(m, 1, 32));
            m = fmaxf(m, __shfl_xor(m, 2, 32));
            m = fmaxf(m, __shfl_xor(m, 4, 32));
            m = fmaxf(m, __shfl_xor(m, 8, 32));
            float nm = fmaxf(row_max[r], m);
            float sc = __expf(row_max[r] - nm);
            row_max[r] = nm;
            float p0 = __expf(v0 - nm);
            float p1 = __expf(v1 - nm);
            float p2 = __expf(v2 - nm);
            float p3 = __expf(v3 - nm);
            float ps = (p0 + p1) + (p2 + p3);
            ps += __shfl_xor(ps, 1, 32);
            ps += __shfl_xor(ps, 2, 32);
            ps += __shfl_xor(ps, 4, 32);
            ps += __shfl_xor(ps, 8, 32);
            row_sum[r] = row_sum[r] * sc + ps;
#pragma unroll
            for (int t = 0; t < 4; ++t) acc[t][r] = acc[t][r] * sc;
            // C-layout -> LDS so P can be reloaded in A-fragment layout
            int prow = (w * 16 + r + 8 * g) * LDT;
            Ps[prow + ln]      = f2bf(p0);
            Ps[prow + 16 + ln] = f2bf(p1);
            Ps[prow + 32 + ln] = f2bf(p2);
            Ps[prow + 48 + ln] = f2bf(p3);
        }

        // ctx += P(16x64) * V(64x64): 2 k-steps x 4 d-tiles
#pragma unroll
        for (int ks = 0; ks < 2; ++ks) {
            Frag pf;
            const unsigned short* pp = &Ps[(w * 16 + ln) * LDT + ks * 32 + g * 8];
            pf.q[0] = *(const uint4*)pp;
            pf.q[1] = *(const uint4*)(pp + 16);
#pragma unroll
            for (int td = 0; td < 4; ++td) {
                Frag vf;
                const unsigned short* vp = &Vt[(16 * td + ln) * LDT + ks * 32 + g * 8];
                vf.q[0] = *(const uint4*)vp;
                vf.q[1] = *(const uint4*)(vp + 16);
                acc[td] = wmma_bf16(pf.v, vf.v, acc[td]);
            }
        }
    }

    // normalize and write ctx (bf16, [B,S,HID] layout for the O-projection)
#pragma unroll
    for (int td = 0; td < 4; ++td)
#pragma unroll
        for (int r = 0; r < 8; ++r) {
            float val = acc[td][r] / row_sum[r];
            int s   = qb + 16 * w + r + 8 * g;
            int col = h * HD + 16 * td + ln;
            ctx[((size_t)b * SEQ + s) * HIDDEN + col] = f2bf(val);
        }
}

// ---------------------------------------------------------------------------
extern "C" void kernel_launch(void* const* d_in, const int* in_sizes, int n_in,
                              void* d_out, int out_size, void* d_ws, size_t ws_size,
                              hipStream_t stream) {
    (void)in_sizes; (void)n_in; (void)out_size; (void)ws_size;
    const float* x    = (const float*)d_in[0];   // hidden_states [B,S,HID]
    const float* mask = (const float*)d_in[1];   // attention_mask [B,S]
    const float* wq   = (const float*)d_in[2];
    const float* wk   = (const float*)d_in[3];
    const float* wv   = (const float*)d_in[4];
    const float* wo   = (const float*)d_in[5];

    const size_t MAT  = (size_t)(BATCH * SEQ) * HIDDEN;   // 8.39M elems
    const size_t WMAT = (size_t)HIDDEN * HIDDEN;          // 4.19M elems
    unsigned short* xb  = (unsigned short*)d_ws;          // bf16 copies
    unsigned short* wqb = xb  + MAT;
    unsigned short* wkb = wqb + WMAT;
    unsigned short* wvb = wkb + WMAT;
    unsigned short* wob = wvb + WMAT;
    unsigned short* qws = wob + WMAT;                     // bf16 [B,H,S,D]
    unsigned short* kws = qws + MAT;
    unsigned short* vws = kws + MAT;
    unsigned short* cws = vws + MAT;                      // bf16 [B,S,HID]

    // one-time f32 -> bf16 conversions (removes convert VALU from GEMM loops)
    cvt_f32_bf16<<<(int)(MAT  / 2048), 256, 0, stream>>>(x,  xb);
    cvt_f32_bf16<<<(int)(WMAT / 2048), 256, 0, stream>>>(wq, wqb);
    cvt_f32_bf16<<<(int)(WMAT / 2048), 256, 0, stream>>>(wk, wkb);
    cvt_f32_bf16<<<(int)(WMAT / 2048), 256, 0, stream>>>(wv, wvb);
    cvt_f32_bf16<<<(int)(WMAT / 2048), 256, 0, stream>>>(wo, wob);

    dim3 ggrid(HIDDEN / 128, (BATCH * SEQ) / 128);        // (16, 32)
    gemm_nt<true><<<ggrid, 256, 0, stream>>>(xb, wqb, qws);
    gemm_nt<true><<<ggrid, 256, 0, stream>>>(xb, wkb, kws);
    gemm_nt<true><<<ggrid, 256, 0, stream>>>(xb, wvb, vws);

    attn_kernel<<<dim3(SEQ / 64, BATCH * NH), 128, 0, stream>>>(qws, kws, vws, mask, cws);

    gemm_nt<false><<<ggrid, 256, 0, stream>>>(cws, wob, d_out);
}
